// SO2_Linear_7945689498244
// MI455X (gfx1250) — matrix operand order repaired and checked by reading, not compiled, not run on hardware
//
#include <hip/hip_runtime.h>
#include <hip/hip_bf16.h>
#include <math.h>

// ---------------------------------------------------------------------------
// SO2_Linear fused kernel for gfx1250 (MI455X).
//   - bf16 WMMA (v_wmma_f32_16x16x32_bf16) everywhere, f32 accumulate
//   - TM=32 edges/block: 2 WMMAs share every B fragment (halves L2 traffic)
//   - radial MLP fused in-block: 1536-wide activation never touches HBM
//   - coefficient permutation fused into loads (gather) and stores (scatter)
// ---------------------------------------------------------------------------

typedef __attribute__((ext_vector_type(16))) __bf16 v16bf;
typedef __attribute__((ext_vector_type(8)))  __bf16 v8bf;
typedef __attribute__((ext_vector_type(8)))  float  v8f;

#define K_COEF 19
#define C_CH   128
#define KC     (K_COEF * C_CH)   // 2432
#define TM     32                // edges per block
#define PAD    8                 // bf16 elements of LDS row padding (16B)
#define LDC    (C_CH + PAD)

// m-major row a  <->  l-major row ORDER[a]
__constant__ int ORDER[19] = {0, 2, 6, 11, 16,   // m=0  (l=0..4)
                              3, 7, 12, 17,      // m=+1 (l=1..4)
                              1, 5, 10, 15,      // m=-1
                              8, 13, 18,         // m=+2 (l=2..4)
                              4, 9, 14};         // m=-2

__device__ __forceinline__ v8f wmma_bf16(v16bf a, v16bf b, v8f c) {
    return __builtin_amdgcn_wmma_f32_16x16x32_bf16(
        /*neg_a=*/false, a, /*neg_b=*/false, b,
        /*c_mod=*/(short)0, c, /*reuse_a=*/false, /*reuse_b=*/false);
}

// A fragment (16x32 bf16, M rows) from LDS row-major [.., ld] tile.
// Lane L: M = L&15; VGPRs 0-3 hold K = base+0..7, VGPRs 4-7 hold K = base+16..23,
// base = (L>=16 ? 8 : 0)  (per ISA 16-bit A-matrix table).
__device__ __forceinline__ v16bf load_frag_a(const __bf16* base, int ld,
                                             int k0, int lane) {
    const int m = lane & 15;
    const int c = (lane >> 4) << 3;            // 0 or 8
    const __bf16* p = base + m * ld + k0 + c;
    v8bf lo = *reinterpret_cast<const v8bf*>(p);
    v8bf hi = *reinterpret_cast<const v8bf*>(p + 16);
    v16bf r;
#pragma unroll
    for (int i = 0; i < 8; ++i) { r[i] = lo[i]; r[8 + i] = hi[i]; }
    return r;
}

// B fragment (32x16 bf16, K x Nout) for D = A @ W^T: B[k, j] = W[j, k].
// Lane L: column j = j0 + (L&15); elements 0-7 hold K = base+0..7, 8-15 hold
// K = base+16..23, base = (L>=16 ? 8 : 0) (matches A fragment K order).
__device__ __forceinline__ v16bf load_frag_b(const __bf16* __restrict__ W,
                                             int din, int j0, int k0, int lane) {
    const int j = j0 + (lane & 15);
    const int c = (lane >> 4) << 3;            // 0 or 8
    const __bf16* p = W + (size_t)j * din + k0 + c;
    v8bf lo = *reinterpret_cast<const v8bf*>(p);
    v8bf hi = *reinterpret_cast<const v8bf*>(p + 16);
    v16bf r;
#pragma unroll
    for (int i = 0; i < 8; ++i) { r[i] = lo[i]; r[8 + i] = hi[i]; }
    return r;
}

// 32x16 output tile: two accumulators (edge rows 0-15 / 16-31) share B frags.
template <int NKT>
__device__ __forceinline__ void gemm_tile32(const __bf16* __restrict__ Abase,
                                            int ldm,
                                            const __bf16* __restrict__ W,
                                            int din, int j0, int lane,
                                            v8f& acc0, v8f& acc1) {
#pragma unroll 4
    for (int kt = 0; kt < NKT; ++kt) {
        const v16bf b = load_frag_b(W, din, j0, kt * 32, lane);
        acc0 = wmma_bf16(load_frag_a(Abase, ldm, kt * 32, lane), b, acc0);
        acc1 = wmma_bf16(load_frag_a(Abase + 16 * ldm, ldm, kt * 32, lane), b,
                         acc1);
    }
}

// One radial layer on a [32,128] tile:
//   dst = silu(layernorm(src @ W^T + bias) * gamma + beta)
// src/dst: LDS bf16 with row stride LDC. hs: LDS f32 [32,128].
__device__ __forceinline__ void dense_ln_silu(
    const __bf16* __restrict__ src, __bf16* __restrict__ dst,
    const __bf16* __restrict__ W, const float* __restrict__ bias,
    const float* __restrict__ gamma, const float* __restrict__ beta,
    float* __restrict__ hs, int wave, int lane) {
    v8f a0 = {0.f, 0.f, 0.f, 0.f, 0.f, 0.f, 0.f, 0.f};
    v8f a1 = a0;
    const int j0 = wave * 16;                    // wave owns 16 output columns
    gemm_tile32<4>(src, LDC, W, C_CH, j0, lane, a0, a1);
    const int j = j0 + (lane & 15);
    const float bv = bias[j];
#pragma unroll
    for (int r = 0; r < 8; ++r) {
        const int m = r + ((lane >> 4) << 3);    // D layout: VGPR r -> row r/r+8
        hs[m * C_CH + j] = a0[r] + bv;
        hs[(m + 16) * C_CH + j] = a1[r] + bv;
    }
    __syncthreads();
    // LayerNorm + SiLU: each wave handles 4 rows, 4 values per lane
#pragma unroll
    for (int rr = 0; rr < 4; ++rr) {
        const int row = wave * 4 + rr;
        float v[4], s1 = 0.f, s2 = 0.f;
#pragma unroll
        for (int q = 0; q < 4; ++q) {
            v[q] = hs[row * C_CH + lane + q * 32];
            s1 += v[q];
            s2 += v[q] * v[q];
        }
#pragma unroll
        for (int off = 16; off; off >>= 1) {
            s1 += __shfl_xor(s1, off, 32);
            s2 += __shfl_xor(s2, off, 32);
        }
        const float mean = s1 * (1.0f / 128.0f);
        const float var  = s2 * (1.0f / 128.0f) - mean * mean;
        const float rstd = rsqrtf(var + 1e-5f);
#pragma unroll
        for (int q = 0; q < 4; ++q) {
            const int c = lane + q * 32;
            const float nv = (v[q] - mean) * rstd * gamma[c] + beta[c];
            const float sv = nv / (1.0f + expf(-nv));
            dst[row * LDC + c] = (__bf16)sv;
        }
    }
    __syncthreads();
}

// One m-pass: (a) compute this pass's slice of the radial output via WMMA and
// FiLM-scale it against permutation-gathered x into xm_s; (b) block GEMM
// against fc weights with permutation scatter-store.
template <int DIN, int NROWS, int NHALF, int RADB, int AB>
__device__ __forceinline__ void so2_pass(
    __bf16* __restrict__ xm_s, const __bf16* __restrict__ h2,
    const __bf16* __restrict__ w3b, const float* __restrict__ b3,
    const __bf16* __restrict__ W, const float* __restrict__ bfc0,
    const float* __restrict__ x, float* __restrict__ out, int n0, int wave,
    int lane) {
    constexpr int LDM = NHALF * DIN + PAD;       // 648 / 1032 / 776

    // (a) radial slice [32, DIN] + FiLM scale
    constexpr int NRADT = DIN / 16;              // 40 / 32 / 24 tiles
#pragma unroll 1
    for (int t = wave; t < NRADT; t += 8) {
        v8f a0 = {0.f, 0.f, 0.f, 0.f, 0.f, 0.f, 0.f, 0.f};
        v8f a1 = a0;
        const int j0 = t * 16;
        gemm_tile32<4>(h2, LDC, w3b, C_CH, RADB + j0, lane, a0, a1);
        const int jl = j0 + (lane & 15);         // col within rad slice
        const int sc = jl >> 7;                  // channel-segment index
        const int c  = jl & 127;
        const float b3v = b3[RADB + jl];
        const int xcP = ORDER[AB + sc] * C_CH + c;
        const int xcM = (NHALF == 2) ? ORDER[AB + NROWS + sc] * C_CH + c : 0;
#pragma unroll
        for (int h = 0; h < 2; ++h) {
            const v8f& acc = h ? a1 : a0;
#pragma unroll
            for (int r = 0; r < 8; ++r) {
                const int m = h * 16 + r + ((lane >> 4) << 3);
                const float* xrow = x + (size_t)(n0 + m) * KC;
                const float radv = acc[r] + b3v;
                xm_s[m * LDM + sc * C_CH + c] = (__bf16)(radv * xrow[xcP]);
                if (NHALF == 2)                  // -m rows share rad slice
                    xm_s[m * LDM + (NROWS + sc) * C_CH + c] =
                        (__bf16)(radv * xrow[xcM]);
            }
        }
    }
    __syncthreads();

    // (b) block GEMM: Y = Ahalf @ W^T (+bias for m=0), permuted scatter-store
    constexpr int TPH   = DIN / 16;              // tiles per half
    constexpr int NTILE = TPH * NHALF;           // 40 / 64 / 48
    constexpr int NKT   = DIN / 32;              // 20 / 16 / 12 k-steps
#pragma unroll 1
    for (int t = wave; t < NTILE; t += 8) {
        const int half = t / TPH;
        const int j0   = (t % TPH) * 16;
        v8f a0 = {0.f, 0.f, 0.f, 0.f, 0.f, 0.f, 0.f, 0.f};
        v8f a1 = a0;
        gemm_tile32<NKT>(xm_s + half * DIN, LDM, W, DIN, j0, lane, a0, a1);
        const int jl = j0 + (lane & 15);
        const int a  = AB + half * NROWS + (jl >> 7);
        const int c  = jl & 127;
        const float bv = (AB == 0) ? bfc0[jl] : 0.0f;
        const int oc = ORDER[a] * C_CH + c;      // inverse permutation
#pragma unroll
        for (int h = 0; h < 2; ++h) {
            const v8f& acc = h ? a1 : a0;
#pragma unroll
            for (int r = 0; r < 8; ++r) {
                const int m = h * 16 + r + ((lane >> 4) << 3);
                out[(size_t)(n0 + m) * KC + oc] = acc[r] + bv;
            }
        }
    }
    __syncthreads();                             // before xm_s is overwritten
}

__global__ void __launch_bounds__(256)
so2_fused(const float* __restrict__ x, const float* __restrict__ xe,
          const __bf16* __restrict__ wfc0, const float* __restrict__ bfc0,
          const __bf16* __restrict__ wfc1, const __bf16* __restrict__ wfc2,
          const __bf16* __restrict__ w1b, const float* __restrict__ b1,
          const float* __restrict__ g1, const float* __restrict__ be1,
          const __bf16* __restrict__ w2b, const float* __restrict__ b2,
          const float* __restrict__ g2, const float* __restrict__ be2,
          const __bf16* __restrict__ w3b, const float* __restrict__ b3,
          float* __restrict__ out) {
    __shared__ __bf16 xm_s[TM * (1024 + PAD)];   // FiLM-scaled A tile (66 KB)
    __shared__ float  hs[TM * C_CH];             // f32 scratch for LN (16 KB)
    __shared__ __bf16 pa[TM * LDC];              // activation ping (8.5 KB)
    __shared__ __bf16 pb[TM * LDC];              // activation pong (8.5 KB)

    const int tid  = threadIdx.x;
    const int wave = tid >> 5;
    const int lane = tid & 31;
    const int n0   = blockIdx.x * TM;

    // ---- stage x_edge tile as bf16 -------------------------------------
    for (int i = tid; i < TM * C_CH; i += 256) {
        const int m = i >> 7, c = i & 127;
        pa[m * LDC + c] = (__bf16)xe[(size_t)(n0 + m) * C_CH + c];
    }
    __syncthreads();

    // ---- radial MLP layers 1 & 2 ---------------------------------------
    dense_ln_silu(pa, pb, w1b, b1, g1, be1, hs, wave, lane);  // pa -> pb
    dense_ln_silu(pb, pa, w2b, b2, g2, be2, hs, wave, lane);  // pb -> pa (h2)

    // ---- three m-passes -------------------------------------------------
    so2_pass<640, 5, 1, 0, 0>(xm_s, pa, w3b, b3, wfc0, bfc0, x, out, n0, wave,
                              lane);
    so2_pass<512, 4, 2, 640, 5>(xm_s, pa, w3b, b3, wfc1, bfc0, x, out, n0,
                                wave, lane);
    so2_pass<384, 3, 2, 1152, 13>(xm_s, pa, w3b, b3, wfc2, bfc0, x, out, n0,
                                  wave, lane);
}

// one-shot fp32 -> bf16 weight conversion into workspace
__global__ void cvt_bf16(const float* __restrict__ s, __bf16* __restrict__ d,
                         int n) {
    const int i = blockIdx.x * blockDim.x + threadIdx.x;
    if (i < n) d[i] = (__bf16)s[i];
}

extern "C" void kernel_launch(void* const* d_in, const int* in_sizes, int n_in,
                              void* d_out, int out_size, void* d_ws,
                              size_t ws_size, hipStream_t stream) {
    const float* x    = (const float*)d_in[0];
    const float* xe   = (const float*)d_in[1];
    const float* fc0w = (const float*)d_in[2];
    const float* fc0b = (const float*)d_in[3];
    const float* fc1w = (const float*)d_in[4];
    const float* fc2w = (const float*)d_in[5];
    const float* rw1  = (const float*)d_in[6];
    const float* rb1  = (const float*)d_in[7];
    const float* rg1  = (const float*)d_in[8];
    const float* rbe1 = (const float*)d_in[9];
    const float* rw2  = (const float*)d_in[10];
    const float* rb2  = (const float*)d_in[11];
    const float* rg2  = (const float*)d_in[12];
    const float* rbe2 = (const float*)d_in[13];
    const float* rw3  = (const float*)d_in[14];
    const float* rb3  = (const float*)d_in[15];
    float* out = (float*)d_out;

    // bf16 weight cache in workspace (~2 MB; L2-resident across all blocks)
    __bf16* wfc0 = (__bf16*)d_ws;
    __bf16* wfc1 = wfc0 + 640 * 640;
    __bf16* wfc2 = wfc1 + 512 * 512;
    __bf16* w1b  = wfc2 + 384 * 384;
    __bf16* w2b  = w1b + 128 * 128;
    __bf16* w3b  = w2b + 128 * 128;

    cvt_bf16<<<(640 * 640 + 255) / 256, 256, 0, stream>>>(fc0w, wfc0, 640 * 640);
    cvt_bf16<<<(512 * 512 + 255) / 256, 256, 0, stream>>>(fc1w, wfc1, 512 * 512);
    cvt_bf16<<<(384 * 384 + 255) / 256, 256, 0, stream>>>(fc2w, wfc2, 384 * 384);
    cvt_bf16<<<(128 * 128 + 255) / 256, 256, 0, stream>>>(rw1, w1b, 128 * 128);
    cvt_bf16<<<(128 * 128 + 255) / 256, 256, 0, stream>>>(rw2, w2b, 128 * 128);
    cvt_bf16<<<(1536 * 128 + 255) / 256, 256, 0, stream>>>(rw3, w3b, 1536 * 128);

    const int n = in_sizes[0] / KC;              // number of edges (32768)
    so2_fused<<<n / TM, 256, 0, stream>>>(x, xe, wfc0, fc0b, wfc1, wfc2,
                                          w1b, rb1, rg1, rbe1,
                                          w2b, rb2, rg2, rbe2,
                                          w3b, rb3, out);
}
